// CombineNode_7395933684091
// MI455X (gfx1250) — matrix operand order repaired and checked by reading, not compiled, not run on hardware
//
#include <hip/hip_runtime.h>
#include <math.h>

// ---------------- types ----------------
typedef __attribute__((ext_vector_type(16))) __bf16         v16bf;
typedef __attribute__((ext_vector_type(8)))  float          v8f;
typedef __attribute__((ext_vector_type(8)))  unsigned short ushort8v;

// ---------------- problem constants ----------------
#define BATCH 8192
#define DIM   1024
#define HID   32
#define NLEAF 128
#define NINT  16
#define CPI   8
// derived
#define KT_LEAF (DIM / 32)      // 32 K-tiles of 32
#define MTILES  (BATCH / 16)    // 512
#define MBLOCKS (BATCH / 64)    // 128 (hidden-gemm: 64 rows per block)
#define MSUPER  (BATCH / 256)   // 32  (leaf-gemm: 256 rows per block)
#define FRAG_US 512             // ushorts per fragment tile (32 lanes * 16)
#define MT_US   (KT_LEAF * FRAG_US) // ushorts per packed-A M-tile (16384)

// ---------------- helpers ----------------
static __device__ __forceinline__ unsigned short f2bf(float f) {
    unsigned u = __builtin_bit_cast(unsigned, f);
    u += 0x7FFFu + ((u >> 16) & 1u);           // round-to-nearest-even
    return (unsigned short)(u >> 16);
}
static __device__ __forceinline__ float bf2f(unsigned short h) {
    return __builtin_bit_cast(float, ((unsigned)h) << 16);
}
static __device__ __forceinline__ v8f wmma_bf16(v16bf a, v16bf b, v8f c) {
    return __builtin_amdgcn_wmma_f32_16x16x32_bf16(false, a, false, b, (short)0, c, false, false);
}

// ======================================================================
// Pack x[B,D] (f32) into bf16 A-fragments.
// CDNA5 16-bit A 16x32 layout: lane L (0-15): row M=L, elems 0..7 -> K 0..7,
// elems 8..15 -> K 16..23; lane L+16: row M=L, K 8..15 and 24..31.
// packedX[(mtile*KT + ktile)*32 + lane][16]
// ======================================================================
__global__ __launch_bounds__(256) void pack_x_kernel(const float* __restrict__ x,
                                                     unsigned short* __restrict__ dst) {
    int gid   = blockIdx.x * 256 + threadIdx.x;       // MTILES*KT_LEAF*32 threads
    int lane  = gid & 31;
    int t     = gid >> 5;
    int ktile = t & (KT_LEAF - 1);
    int mtile = t >> 5;
    int row   = mtile * 16 + (lane & 15);
    int kbase = ktile * 32 + ((lane >> 4) << 3);
    const float* xr = x + (size_t)row * DIM + kbase;
    unsigned short* o = dst + (size_t)gid * 16;
#pragma unroll
    for (int e = 0; e < 16; ++e) {
        int k = (e & 7) + ((e >> 3) << 4);            // 0..7, then +16
        o[e] = f2bf(xr[k]);
    }
}

// ======================================================================
// Pack per-term weight [K,32] (f32, row-major, term stride K*32) into bf16
// B-fragments. 16-bit B 32x16 layout: col N = lane&15; lanes 0-15 hold
// K 0..7,16..23, lanes 16-31 hold K 8..15,24..31.
// dst[((term*2 + ntile)*KT + ktile)*32 + lane][16]
// ======================================================================
__global__ __launch_bounds__(256) void pack_w_kernel(const float* __restrict__ src,
                                                     unsigned short* __restrict__ dst,
                                                     int KT, int K, int total) {
    int gid = blockIdx.x * 256 + threadIdx.x;
    if (gid >= total) return;
    int lane  = gid & 31;
    int t     = gid >> 5;
    int ktile = t % KT;
    int t2    = t / KT;
    int ntile = t2 & 1;
    int term  = t2 >> 1;
    int n  = ntile * 16 + (lane & 15);
    int kb = ktile * 32 + ((lane >> 4) << 3);
    const float* s = src + (size_t)term * K * HID;
    unsigned short* o = dst + (size_t)gid * 16;
#pragma unroll
    for (int e = 0; e < 16; ++e)
        o[e] = f2bf(s[(size_t)(kb + (e & 7) + ((e >> 3) << 4)) * HID + n]);
}

// ======================================================================
// Leaf GEMM: leaf_h = tanh(x @ W_l + b_l), packed-A + packed-B inputs.
// grid.x = NLEAF * MSUPER ; 256 threads = 8 waves.
// Each wave owns a 64x16 C tile (4 M-frags x 1 N-frag) -> one B-fragment
// feeds 4 WMMAs. Explicit register double-buffering: iteration kt+1's five
// fragments are loaded into a rotating second register set before issuing
// iteration kt's WMMAs, so the wave overlaps loads with matrix ops instead
// of s_wait_loadcnt 0 before every WMMA.
// ======================================================================
__global__ __launch_bounds__(256) void leaf_gemm_kernel(const unsigned short* __restrict__ pX,
                                                        const unsigned short* __restrict__ pW,
                                                        const float* __restrict__ bias,
                                                        unsigned short* __restrict__ lh) {
    int l      = blockIdx.x & (NLEAF - 1);
    int msuper = blockIdx.x >> 7;                     // 0..31 (256 rows each)
    int lane   = threadIdx.x & 31;
    int wave   = threadIdx.x >> 5;                    // 0..7
    int ntile  = wave & 1;
    int mtile0 = msuper * 16 + (wave >> 1) * 4;       // 4 consecutive M-tiles

    const unsigned short* aP = pX + ((size_t)mtile0 * KT_LEAF * 32 + lane) * 16;
    const unsigned short* bP = pW + (((size_t)(l * 2 + ntile) * KT_LEAF) * 32 + lane) * 16;

    v8f acc0 = {}, acc1 = {}, acc2 = {}, acc3 = {};

    // prologue: fragment set for kt = 0
    v16bf b  = *(const v16bf*)(bP);
    v16bf a0 = *(const v16bf*)(aP);
    v16bf a1 = *(const v16bf*)(aP + 1 * MT_US);
    v16bf a2 = *(const v16bf*)(aP + 2 * MT_US);
    v16bf a3 = *(const v16bf*)(aP + 3 * MT_US);

#pragma unroll 1
    for (int kt = 0; kt < KT_LEAF - 1; ++kt) {
        const unsigned short* aN = aP + (size_t)(kt + 1) * FRAG_US;
        const unsigned short* bN = bP + (size_t)(kt + 1) * FRAG_US;
        // prefetch next iteration's fragments into a distinct register set
        v16bf nb  = *(const v16bf*)(bN);
        v16bf na0 = *(const v16bf*)(aN);
        v16bf na1 = *(const v16bf*)(aN + 1 * MT_US);
        v16bf na2 = *(const v16bf*)(aN + 2 * MT_US);
        v16bf na3 = *(const v16bf*)(aN + 3 * MT_US);
        // consume current set
        acc0 = wmma_bf16(a0, b, acc0);
        acc1 = wmma_bf16(a1, b, acc1);
        acc2 = wmma_bf16(a2, b, acc2);
        acc3 = wmma_bf16(a3, b, acc3);
        // rotate
        b = nb; a0 = na0; a1 = na1; a2 = na2; a3 = na3;
    }
    // epilogue: last k-tile
    acc0 = wmma_bf16(a0, b, acc0);
    acc1 = wmma_bf16(a1, b, acc1);
    acc2 = wmma_bf16(a2, b, acc2);
    acc3 = wmma_bf16(a3, b, acc3);

    int col  = l * HID + ntile * 16 + (lane & 15);
    float bv = bias[l * HID + ntile * 16 + (lane & 15)];
    int mOff = ((lane >> 4) << 3);                    // C layout: hi half-wave -> M+8
    v8f accs[4] = {acc0, acc1, acc2, acc3};
#pragma unroll
    for (int i = 0; i < 4; ++i) {
        int rowB = (mtile0 + i) * 16 + mOff;
#pragma unroll
        for (int r = 0; r < 8; ++r) {
            float hv = tanhf(accs[i][r] + bv);
            lh[(size_t)(rowB + r) * (NLEAF * HID) + col] = f2bf(hv);
        }
    }
}

// ======================================================================
// Generic hidden GEMM (internal + root): A gathered from bf16 hidden matrix
// [B, lda], per-term column window colStride; B from packed fragments.
// out[B, ldo] bf16, bias per (term,h). grid.x = nterms * MBLOCKS.
// ======================================================================
__global__ __launch_bounds__(256) void hidden_gemm_kernel(const unsigned short* __restrict__ A,
                                                          int lda, int colStride,
                                                          const unsigned short* __restrict__ pW,
                                                          const float* __restrict__ bias,
                                                          unsigned short* __restrict__ out,
                                                          int ldo, int KT, int nterms) {
    int term   = blockIdx.x % nterms;
    int mblock = blockIdx.x / nterms;
    int lane   = threadIdx.x & 31;
    int wave   = threadIdx.x >> 5;
    int mtile  = mblock * 4 + (wave >> 1);
    int ntile  = wave & 1;

    int rowA = mtile * 16 + (lane & 15);
    const unsigned short* aRow = A + (size_t)rowA * lda + term * colStride + ((lane >> 4) << 3);
    const unsigned short* bP   = pW + (((size_t)(term * 2 + ntile) * KT) * 32 + lane) * 16;

    v8f acc = {};
#pragma unroll 4
    for (int kt = 0; kt < KT; ++kt) {
        const unsigned short* p = aRow + kt * 32;
        union { v16bf v; ushort8v h[2]; } au;
        au.h[0] = *(const ushort8v*)(p);              // K kbase..kbase+7
        au.h[1] = *(const ushort8v*)(p + 16);         // K kbase+16..kbase+23
        v16bf b = *(const v16bf*)(bP + (size_t)kt * FRAG_US);
        acc = wmma_bf16(au.v, b, acc);
    }

    int col  = term * HID + ntile * 16 + (lane & 15);
    float bv = bias[term * HID + ntile * 16 + (lane & 15)];
    int rowB = mtile * 16 + ((lane >> 4) << 3);
#pragma unroll
    for (int r = 0; r < 8; ++r) {
        float hv = tanhf(acc[r] + bv);
        out[(size_t)(rowB + r) * ldo + col] = f2bf(hv);
    }
}

// ======================================================================
// Predict heads: out[t*B + b] = tanh(dot(h[b, t*32 : t*32+32], Wp[t]) + bp[t])
// ======================================================================
__global__ __launch_bounds__(256) void predict_kernel(const unsigned short* __restrict__ h,
                                                      int ldh,
                                                      const float* __restrict__ Wp,
                                                      const float* __restrict__ bp,
                                                      float* __restrict__ out,
                                                      int total) {
    int gid = blockIdx.x * 256 + threadIdx.x;
    if (gid >= total) return;
    int b = gid & (BATCH - 1);
    int t = gid >> 13;                                 // BATCH == 2^13
    const unsigned short* hp = h + (size_t)b * ldh + t * HID;
    const float* wp = Wp + t * HID;
    float s = bp[t];
#pragma unroll
    for (int j = 0; j < HID; ++j)
        s += bf2f(hp[j]) * wp[j];
    out[gid] = tanhf(s);
}

// ======================================================================
// Launch
// ======================================================================
extern "C" void kernel_launch(void* const* d_in, const int* in_sizes, int n_in,
                              void* d_out, int out_size, void* d_ws, size_t ws_size,
                              hipStream_t stream) {
    (void)in_sizes; (void)n_in; (void)out_size; (void)ws_size;

    const float* x       = (const float*)d_in[0];
    const float* leaf_W  = (const float*)d_in[1];
    const float* leaf_b  = (const float*)d_in[2];
    const float* int_W   = (const float*)d_in[3];
    const float* int_b   = (const float*)d_in[4];
    const float* root_W  = (const float*)d_in[5];
    const float* root_b  = (const float*)d_in[6];
    const float* leaf_Wp = (const float*)d_in[7];
    const float* leaf_bp = (const float*)d_in[8];
    const float* int_Wp  = (const float*)d_in[9];
    const float* int_bp  = (const float*)d_in[10];
    const float* root_Wp = (const float*)d_in[11];
    const float* root_bp = (const float*)d_in[12];
    float* out = (float*)d_out;

    // ---- workspace carve-up (all sizes 256B-aligned) ----
    char* ws = (char*)d_ws;
    size_t off = 0;
    auto carve = [&](size_t bytes) { void* p = ws + off; off += (bytes + 255) & ~(size_t)255; return p; };
    unsigned short* pX  = (unsigned short*)carve((size_t)MTILES * KT_LEAF * 32 * 16 * 2); // 16 MB
    unsigned short* pLW = (unsigned short*)carve((size_t)NLEAF * 2 * KT_LEAF * 32 * 16 * 2); // 8 MB
    unsigned short* pIW = (unsigned short*)carve((size_t)NINT  * 2 * 8  * 32 * 16 * 2);   // 256 KB
    unsigned short* pRW = (unsigned short*)carve((size_t)1     * 2 * 16 * 32 * 16 * 2);   // 32 KB
    unsigned short* lh  = (unsigned short*)carve((size_t)BATCH * NLEAF * HID * 2);        // 64 MB
    unsigned short* ih  = (unsigned short*)carve((size_t)BATCH * NINT  * HID * 2);        // 8 MB
    unsigned short* rh  = (unsigned short*)carve((size_t)BATCH * HID * 2);                // 512 KB

    // ---- packing (independent) ----
    pack_x_kernel<<<(MTILES * KT_LEAF * 32) / 256, 256, 0, stream>>>(x, pX);
    {
        int total = NLEAF * 2 * KT_LEAF * 32;  // 262144
        pack_w_kernel<<<(total + 255) / 256, 256, 0, stream>>>(leaf_W, pLW, KT_LEAF, DIM, total);
    }
    {
        int total = NINT * 2 * 8 * 32;         // 8192   (K = CPI*HID = 256)
        pack_w_kernel<<<(total + 255) / 256, 256, 0, stream>>>(int_W, pIW, 8, CPI * HID, total);
    }
    {
        int total = 1 * 2 * 16 * 32;           // 1024   (K = NINT*HID = 512)
        pack_w_kernel<<<(total + 255) / 256, 256, 0, stream>>>(root_W, pRW, 16, NINT * HID, total);
    }

    // ---- level 0: leaves ----
    leaf_gemm_kernel<<<NLEAF * MSUPER, 256, 0, stream>>>(pX, pLW, leaf_b, lh);
    {
        int total = NLEAF * BATCH;             // 1048576
        predict_kernel<<<(total + 255) / 256, 256, 0, stream>>>(lh, NLEAF * HID, leaf_Wp, leaf_bp,
                                                                out, total);
    }

    // ---- level 1: internals (A = contiguous 256-col windows of lh) ----
    hidden_gemm_kernel<<<NINT * MBLOCKS, 256, 0, stream>>>(lh, NLEAF * HID, CPI * HID,
                                                           pIW, int_b, ih, NINT * HID, 8, NINT);
    {
        int total = NINT * BATCH;              // 131072
        predict_kernel<<<(total + 255) / 256, 256, 0, stream>>>(ih, NINT * HID, int_Wp, int_bp,
                                                                out + (size_t)NLEAF * BATCH, total);
    }

    // ---- level 2: root ----
    hidden_gemm_kernel<<<MBLOCKS, 256, 0, stream>>>(ih, NINT * HID, 0,
                                                    pRW, root_b, rh, HID, 16, 1);
    {
        int total = BATCH;                     // 8192
        predict_kernel<<<(total + 255) / 256, 256, 0, stream>>>(rh, HID, root_Wp, root_bp,
                                                                out + (size_t)(NLEAF + NINT) * BATCH, total);
    }
}
